// Attention_65214783422545
// MI455X (gfx1250) — matrix operand order repaired and checked by reading, not compiled, not run on hardware
//
#include <hip/hip_runtime.h>

// ---------------------------------------------------------------------------
// Types for CDNA5 WMMA (wave32): v_wmma_f32_16x16x32_bf16
// ---------------------------------------------------------------------------
typedef __attribute__((ext_vector_type(16))) __bf16        v16bf;
typedef __attribute__((ext_vector_type(8)))  __bf16        v8bf;
typedef __attribute__((ext_vector_type(8)))  float         v8f;
typedef __attribute__((ext_vector_type(8)))  unsigned int  v8u;

#define DEV __device__ __forceinline__

DEV unsigned short f2bf(float f) {            // f32 -> bf16, round-nearest-even
  unsigned u = __float_as_uint(f);
  u += 0x7FFFu + ((u >> 16) & 1u);
  return (unsigned short)(u >> 16);
}
DEV unsigned pack2(float lo, float hi) {
  return (unsigned)f2bf(lo) | ((unsigned)f2bf(hi) << 16);
}
DEV v16bf make_afrag(v8bf lo, v8bf hi) {      // concat two 16B chunks
  return __builtin_shufflevector(lo, hi, 0,1,2,3,4,5,6,7,8,9,10,11,12,13,14,15);
}
DEV v8f wmma_bf16(v16bf a, v16bf b, v8f c) {
  return __builtin_amdgcn_wmma_f32_16x16x32_bf16(false, a, false, b,
                                                 (short)0, c, false, false);
}

// Problem constants
#define BB 4
#define SS 1024
#define DD 1024
#define HH 16
#define DK 64

// ---------------------------------------------------------------------------
// f32 -> bf16 conversion (vectorized x4)
// ---------------------------------------------------------------------------
__global__ void cvt_f32_bf16(const float* __restrict__ in,
                             unsigned short* __restrict__ out, int n) {
  int i = (blockIdx.x * blockDim.x + threadIdx.x) * 4;
  if (i < n) {
    float4 f = *(const float4*)(in + i);
    unsigned lo = pack2(f.x, f.y), hi = pack2(f.z, f.w);
    *(uint2*)(out + i) = make_uint2(lo, hi);
  }
}

// ---------------------------------------------------------------------------
// GEMM: C[4096,1024] = A[4096,1024] @ W^T + bias,  A,W bf16 row-major [*,K]
// Block = 128 threads (4 waves); block tile 64x64; wave tile 16x64.
// B (weight) tile is shared by all 4 waves -> staged through LDS with
// global_load_async_to_lds_b128 (ASYNCcnt), double-buffered, padded rows
// (72 elems) for conflict-free ds_load_b128 fragment reads.
// MODE 0: store bf16 row-major (Q,K)   MODE 1: store V transposed per head
// MODE 2: store f32 row-major (final y)
// ---------------------------------------------------------------------------
template <int MODE>
__global__ void gemm4096x1024(const unsigned short* __restrict__ A,
                              const unsigned short* __restrict__ Bw,
                              const float* __restrict__ bias,
                              void* __restrict__ Cout) {
  constexpr int K      = DD;
  constexpr int KSTEP  = 64;          // k per stage
  constexpr int NSTEP  = K / KSTEP;   // 16 stages
  constexpr int LROW   = 72;          // padded LDS row (bf16 elems)
  __shared__ __align__(32) unsigned short bsm[2][64 * LROW];   // 2 x 9 KB

  int blk  = blockIdx.x;              // 1024 blocks: 64 (M) x 16 (N)
  int tm   = blk >> 4, tn = blk & 15;
  int tid  = threadIdx.x;
  int w    = __builtin_amdgcn_readfirstlane((int)(tid >> 5));
  int lane = tid & 31;
  int half = lane >> 4, ln = lane & 15;

  // Async-copy addressing: 64x64 bf16 tile = 512 chunks of 16B; 4 chunks/thread.
  unsigned ldsoff[4];                 // LDS byte offset within one buffer
  unsigned goff[4];                   // global byte offset (k0 = 0)
#pragma unroll
  for (int c = 0; c < 4; ++c) {
    int e  = (tid + c * 128) * 8;     // element index in 64(N) x 64(K) tile
    int nl = e >> 6, kl = e & 63;
    ldsoff[c] = (unsigned)((nl * LROW + kl) * 2);
    goff[c]   = (unsigned)(((tn * 64 + nl) * K + kl) * 2);
  }
  unsigned long long bptr = (unsigned long long)Bw;
  unsigned lds_base = (unsigned)(unsigned long long)(&bsm[0][0]);

  int m = tm * 64 + w * 16 + ln;
  const unsigned short* pa = A + m * K + 16 * half;
  v8f acc[4] = {};

  // prologue: stage 0
#pragma unroll
  for (int c = 0; c < 4; ++c) {
    unsigned dst = lds_base + ldsoff[c];
    unsigned src = goff[c];
    asm volatile("global_load_async_to_lds_b128 %0, %1, %2"
                 :: "v"(dst), "v"(src), "s"(bptr) : "memory");
  }

  for (int i = 0; i < NSTEP; ++i) {
    if (i + 1 < NSTEP) {              // prefetch next stage into other buffer
      unsigned bufoff = (unsigned)(((i + 1) & 1) * 64 * LROW * 2);
      unsigned kadd   = (unsigned)((i + 1) * KSTEP * 2);
#pragma unroll
      for (int c = 0; c < 4; ++c) {
        unsigned dst = lds_base + bufoff + ldsoff[c];
        unsigned src = goff[c] + kadd;
        asm volatile("global_load_async_to_lds_b128 %0, %1, %2"
                     :: "v"(dst), "v"(src), "s"(bptr) : "memory");
      }
      asm volatile("s_wait_asynccnt 0x4" ::: "memory");   // stage i landed
    } else {
      asm volatile("s_wait_asynccnt 0x0" ::: "memory");
    }
    __syncthreads();                  // stage i visible to all waves

    const unsigned short* bs = &bsm[i & 1][0];
#pragma unroll
    for (int kk = 0; kk < 2; ++kk) {  // two 32-wide k sub-steps per stage
      int k0 = i * KSTEP + kk * 32;
      v8bf a0 = *(const v8bf*)(pa + k0);
      v8bf a1 = *(const v8bf*)(pa + k0 + 8);
      v16bf av = make_afrag(a0, a1);
#pragma unroll
      for (int j = 0; j < 4; ++j) {
        const unsigned short* bp =
            bs + (j * 16 + ln) * LROW + kk * 32 + 16 * half;
        v16bf bv = *(const v16bf*)bp;
        acc[j] = wmma_bf16(av, bv, acc[j]);
      }
    }
    __syncthreads();                  // all reads of buf (i&1) done
  }

#pragma unroll
  for (int j = 0; j < 4; ++j) {
    int n = tn * 64 + j * 16 + ln;
    float bb = bias[n];
#pragma unroll
    for (int v = 0; v < 8; ++v) {
      float cv  = acc[j][v] + bb;
      int   row = tm * 64 + w * 16 + v + 8 * half;    // C layout: M=v+8*half
      if (MODE == 0) {
        ((unsigned short*)Cout)[row * DD + n] = f2bf(cv);
      } else if (MODE == 1) {                         // Vt[b][h][d][s]
        int b = row >> 10, s = row & 1023, h = n >> 6, d = n & 63;
        ((unsigned short*)Cout)[(((b * HH + h) * DK + d) << 10) + s] = f2bf(cv);
      } else {
        ((float*)Cout)[row * DD + n] = cv;
      }
    }
  }
}

// ---------------------------------------------------------------------------
// Fused scores + causal softmax.
// Block = 256 threads (8 waves) per (b, h, row-tile rt of 16 rows).
// Wave w owns column tiles ct = w, w+8, ..., 63 (<= rt computed, > rt zeroed).
// Score tiles stay in registers (8x v8f); per-element causal masking only on
// the diagonal tile (wave-uniform branch). Cross-lane max/sum via shfl_xor,
// cross-wave via tiny LDS. Writes normalized attn_w (f32) to d_out.
// ---------------------------------------------------------------------------
__global__ void attn_scores_softmax(const unsigned short* __restrict__ Qb,
                                    const unsigned short* __restrict__ Kb,
                                    float* __restrict__ attn) {
  int unit = blockIdx.x;                     // 4096 = B*H*64
  int rt = unit & 63, h = (unit >> 6) & 15, b = unit >> 10;
  int w    = __builtin_amdgcn_readfirstlane((int)(threadIdx.x >> 5));
  int lane = threadIdx.x & 31;
  int half = lane >> 4, ln = lane & 15;

  const unsigned short* qa =
      Qb + ((b * SS + rt * 16 + ln) * DD + h * DK) + 16 * half;
  const unsigned short* kbase = Kb + (b * SS) * DD + h * DK + 16 * half;

  // A fragments for this row-tile (K = 64 -> two k-steps), shared by all ct
  v16bf av[2];
#pragma unroll
  for (int kk = 0; kk < 2; ++kk) {
    v8bf a0 = *(const v8bf*)(qa + kk * 32);
    v8bf a1 = *(const v8bf*)(qa + kk * 32 + 8);
    av[kk] = make_afrag(a0, a1);
  }

  const float NEG_INF = -__builtin_inff();
  v8f   acc[8];
  float mpart[8];
#pragma unroll
  for (int v = 0; v < 8; ++v) mpart[v] = NEG_INF;
  v8f zero = {0.f, 0.f, 0.f, 0.f, 0.f, 0.f, 0.f, 0.f};

#pragma unroll
  for (int t = 0; t < 8; ++t) {
    int ct = w + t * 8;                      // wave-uniform
    if (ct <= rt) {
      const unsigned short* kb = kbase + (ct * 16 + ln) * DD;
      v8f c = zero;
#pragma unroll
      for (int kk = 0; kk < 2; ++kk) {
        v16bf bv = *(const v16bf*)(kb + kk * 32);
        c = wmma_bf16(av[kk], bv, c);
      }
      if (ct == rt) {                        // diagonal tile: per-element mask
#pragma unroll
        for (int v = 0; v < 8; ++v) {
          float sv = c[v] * 0.125f;          // 1/sqrt(64)
          sv = (ln > v + 8 * half) ? NEG_INF : sv;
          c[v] = sv;
          mpart[v] = fmaxf(mpart[v], sv);
        }
      } else {                               // strictly-lower tile: no mask
#pragma unroll
        for (int v = 0; v < 8; ++v) {
          float sv = c[v] * 0.125f;
          c[v] = sv;
          mpart[v] = fmaxf(mpart[v], sv);
        }
      }
      acc[t] = c;
    }
  }

  // ---- row max: 16-lane butterfly, then across 8 waves via LDS ----
#pragma unroll
  for (int v = 0; v < 8; ++v) {
    float mv = mpart[v];
#pragma unroll
    for (int off = 1; off < 16; off <<= 1) mv = fmaxf(mv, __shfl_xor(mv, off, 32));
    mpart[v] = mv;
  }
  __shared__ float red[8][16];
  __shared__ float stat[16];
  if (ln == 0) {
#pragma unroll
    for (int v = 0; v < 8; ++v) red[w][v + 8 * half] = mpart[v];
  }
  __syncthreads();
  if (threadIdx.x < 16) {
    float m = NEG_INF;
#pragma unroll
    for (int ww = 0; ww < 8; ++ww) m = fmaxf(m, red[ww][threadIdx.x]);
    stat[threadIdx.x] = m;
  }
  __syncthreads();
  float rmax[8];
#pragma unroll
  for (int v = 0; v < 8; ++v) rmax[v] = stat[v + 8 * half];

  // ---- exp + row sum ----
  float spart[8] = {0.f, 0.f, 0.f, 0.f, 0.f, 0.f, 0.f, 0.f};
#pragma unroll
  for (int t = 0; t < 8; ++t) {
    int ct = w + t * 8;
    if (ct <= rt) {
      v8f c = acc[t];
#pragma unroll
      for (int v = 0; v < 8; ++v) {
        float e = __expf(c[v] - rmax[v]);
        c[v] = e;
        spart[v] += e;
      }
      acc[t] = c;
    }
  }
#pragma unroll
  for (int v = 0; v < 8; ++v) {
    float sv = spart[v];
#pragma unroll
    for (int off = 1; off < 16; off <<= 1) sv += __shfl_xor(sv, off, 32);
    spart[v] = sv;
  }
  __syncthreads();                            // stat(max) reads done
  if (ln == 0) {
#pragma unroll
    for (int v = 0; v < 8; ++v) red[w][v + 8 * half] = spart[v];
  }
  __syncthreads();
  if (threadIdx.x < 16) {
    float s = 0.f;
#pragma unroll
    for (int ww = 0; ww < 8; ++ww) s += red[ww][threadIdx.x];
    stat[threadIdx.x] = 1.0f / s;
  }
  __syncthreads();
  float rinv[8];
#pragma unroll
  for (int v = 0; v < 8; ++v) rinv[v] = stat[v + 8 * half];

  // ---- write attn_w: normalized probs, zeros in masked tile region ----
  size_t obase = ((size_t)((b * HH + h) * SS + rt * 16)) * SS;
#pragma unroll
  for (int t = 0; t < 8; ++t) {
    int ct  = w + t * 8;
    int col = ct * 16 + ln;
    if (ct <= rt) {
#pragma unroll
      for (int v = 0; v < 8; ++v)
        attn[obase + (size_t)(v + 8 * half) * SS + col] = acc[t][v] * rinv[v];
    } else {
#pragma unroll
      for (int v = 0; v < 8; ++v)
        attn[obase + (size_t)(v + 8 * half) * SS + col] = 0.0f;
    }
  }
}

// ---------------------------------------------------------------------------
// PV: Yh[b, s, h*64+d] = P[b,h,s,:] @ V  (P read f32 from d_out, cvt to bf16;
// V pre-transposed per head so B fragments are contiguous 32B loads).
// One wave per (b,h,rt); causal: loop only over valid k (padded cols are 0).
// ---------------------------------------------------------------------------
__global__ void attn_pv(const float* __restrict__ attn,
                        const unsigned short* __restrict__ Vt,
                        unsigned short* __restrict__ Yh) {
  int w    = __builtin_amdgcn_readfirstlane((int)(threadIdx.x >> 5));
  int unit = blockIdx.x * 8 + w;             // 4096 wave-units
  int rt = unit & 63, h = (unit >> 6) & 15, b = unit >> 10;
  int lane = threadIdx.x & 31;
  int half = lane >> 4, ln = lane & 15;

  const float* pa =
      attn + ((size_t)((b * HH + h) * SS + rt * 16 + ln)) * SS + 16 * half;
  const unsigned short* vb = Vt + ((b * HH + h) * DK + ln) * SS + 16 * half;

  v8f acc[4] = {};
  int kend = (((rt + 1) * 16) + 31) & ~31;   // wave-uniform
  for (int k0 = 0; k0 < kend; k0 += 32) {
    const float* p = pa + k0;
    float4 c0 = *(const float4*)(p);
    float4 c1 = *(const float4*)(p + 4);
    float4 c2 = *(const float4*)(p + 8);
    float4 c3 = *(const float4*)(p + 12);
    union { v8u u; v16bf b; } af;
    af.u[0] = pack2(c0.x, c0.y); af.u[1] = pack2(c0.z, c0.w);
    af.u[2] = pack2(c1.x, c1.y); af.u[3] = pack2(c1.z, c1.w);
    af.u[4] = pack2(c2.x, c2.y); af.u[5] = pack2(c2.z, c2.w);
    af.u[6] = pack2(c3.x, c3.y); af.u[7] = pack2(c3.z, c3.w);
#pragma unroll
    for (int j = 0; j < 4; ++j) {
      v16bf bv = *(const v16bf*)(vb + j * 16 * SS + k0);
      acc[j] = wmma_bf16(af.b, bv, acc[j]);
    }
  }
#pragma unroll
  for (int j = 0; j < 4; ++j) {
#pragma unroll
    for (int v = 0; v < 8; ++v) {
      int s = rt * 16 + v + 8 * half;
      Yh[(b * SS + s) * DD + h * DK + j * 16 + ln] = f2bf(acc[j][v]);
    }
  }
}

// ---------------------------------------------------------------------------
// Host launch
// ---------------------------------------------------------------------------
extern "C" void kernel_launch(void* const* d_in, const int* in_sizes, int n_in,
                              void* d_out, int out_size, void* d_ws, size_t ws_size,
                              hipStream_t stream) {
  const float* x  = (const float*)d_in[0];
  const float* Wq = (const float*)d_in[1];
  const float* bq = (const float*)d_in[2];
  const float* Wk = (const float*)d_in[3];
  const float* bk = (const float*)d_in[4];
  const float* Wv = (const float*)d_in[5];
  const float* bv = (const float*)d_in[6];
  const float* Wo = (const float*)d_in[7];
  const float* bo = (const float*)d_in[8];

  char* ws = (char*)d_ws;
  const size_t MB = 1024u * 1024u;
  unsigned short* xb  = (unsigned short*)(ws);             // 8 MB  [4096,1024]
  unsigned short* wqb = (unsigned short*)(ws +  8 * MB);   // 2 MB
  unsigned short* wkb = (unsigned short*)(ws + 10 * MB);   // 2 MB
  unsigned short* wvb = (unsigned short*)(ws + 12 * MB);   // 2 MB
  unsigned short* wob = (unsigned short*)(ws + 14 * MB);   // 2 MB
  unsigned short* Qb  = (unsigned short*)(ws + 16 * MB);   // 8 MB
  unsigned short* Kb  = (unsigned short*)(ws + 24 * MB);   // 8 MB
  unsigned short* Vt  = (unsigned short*)(ws + 32 * MB);   // 8 MB  [B,H,64,S]
  unsigned short* Yh  = xb;                                // alias: xb dead by then

  float* attn = (float*)d_out;                             // [B,H,S,S]
  float* yout = attn + (size_t)BB * HH * SS * SS;          // [B,S,D]

  // bf16 conversions
  cvt_f32_bf16<<<(BB * SS * DD / 4 + 255) / 256, 256, 0, stream>>>(x, xb, BB * SS * DD);
  cvt_f32_bf16<<<(DD * DD / 4 + 255) / 256, 256, 0, stream>>>(Wq, wqb, DD * DD);
  cvt_f32_bf16<<<(DD * DD / 4 + 255) / 256, 256, 0, stream>>>(Wk, wkb, DD * DD);
  cvt_f32_bf16<<<(DD * DD / 4 + 255) / 256, 256, 0, stream>>>(Wv, wvb, DD * DD);
  cvt_f32_bf16<<<(DD * DD / 4 + 255) / 256, 256, 0, stream>>>(Wo, wob, DD * DD);

  // QKV projections
  gemm4096x1024<0><<<1024, 128, 0, stream>>>(xb, wqb, bq, (void*)Qb);
  gemm4096x1024<0><<<1024, 128, 0, stream>>>(xb, wkb, bk, (void*)Kb);
  gemm4096x1024<1><<<1024, 128, 0, stream>>>(xb, wvb, bv, (void*)Vt);

  // scores + causal softmax -> attn_w (f32)
  attn_scores_softmax<<<BB * HH * 64, 256, 0, stream>>>(Qb, Kb, attn);

  // P @ V -> Yh (bf16)
  attn_pv<<<BB * HH * 64 / 8, 256, 0, stream>>>(attn, Vt, Yh);

  // output projection -> y (f32)
  gemm4096x1024<2><<<1024, 128, 0, stream>>>(Yh, wob, bo, (void*)yout);
}
